// MPNN_ENN_K_Sum_13039520710679
// MI455X (gfx1250) — matrix operand order repaired and checked by reading, not compiled, not run on hardware
//
#include <hip/hip_runtime.h>
#include <hip/hip_bf16.h>

// ---------------------------------------------------------------------------
// MPNN-ENN fused for MI455X (gfx1250, wave32, WMMA).
// Key idea: never materialize edge_data [E,64,64] (1.6 GB). Instead
//   msg[e][i] = sum_{j,k} We2[i*64+j][k] * eh[e][k] * x[src[e]][j]
// is computed per 16-edge tile as a [16 x 4096] @ [4096 x 64] GEMM whose
// A-operand (outer product x_j * eh_k) is built on the fly in VGPRs and fed
// to v_wmma_f32_16x16x32_f16. The B-operand (We2 permuted) is precomputed in
// exact per-lane fragment order -> contiguous coalesced loads, L2-resident.
// ---------------------------------------------------------------------------

typedef __attribute__((ext_vector_type(16))) _Float16 v16h;
typedef __attribute__((ext_vector_type(8)))  _Float16 v8h;
typedef __attribute__((ext_vector_type(8)))  float    v8f;

constexpr int N_NODES  = 50000;
constexpr int N_EDGES  = 100000;
constexpr int N_GRAPHS = 512;
constexpr int H        = 64;
constexpr int F_NODE   = 16;
constexpr int F_EDGE   = 8;
constexpr int T_STEPS  = 3;
constexpr int N_TILES  = N_EDGES / 16;   // 6250

static __device__ __forceinline__ v8h splat8(_Float16 s) {
  v8h v = {s, s, s, s, s, s, s, s};
  return v;
}
static __device__ __forceinline__ v16h cat8(v8h lo, v8h hi) {
  v16h r;
#pragma unroll
  for (int t = 0; t < 8; ++t) { r[t] = lo[t]; r[t + 8] = hi[t]; }
  return r;
}

// --- Precompute: We2 [4096,64] (row = i*64+j, col = k) -> per-lane WMMA
// B-fragment order: wfrag[((c*4+nb)*32 + lane)*16 + t], where for chunk c the
// GEMM K-index is jk = c*32 + K, K = (t&8)*2 + (lane<16?0:8) + (t&7), i = nb*16 + (lane&15),
// j = jk/64, k = jk%64.
__global__ void wfrag_kernel(const float* __restrict__ We2,
                             _Float16* __restrict__ wfrag) {
  int idx = blockIdx.x * blockDim.x + threadIdx.x;
  if (idx >= 128 * 4 * 32 * 16) return;
  int t  = idx & 15;
  int l  = (idx >> 4) & 31;
  int nb = (idx >> 9) & 3;
  int c  = idx >> 11;
  int i  = nb * 16 + (l & 15);
  int K  = ((t & 8) << 1) + ((l < 16) ? 0 : 8) + (t & 7);
  int jk = c * 32 + K;
  int j  = jk >> 6;
  int k  = jk & 63;
  wfrag[idx] = (_Float16)We2[(size_t)(i * 64 + j) * 64 + k];
}

// --- be2 [4096] reshaped as Bc[i][j]; B-fragments for the 2 bias K-chunks
// (K dim = j, A-operand = x itself).
__global__ void bcfrag_kernel(const float* __restrict__ be2,
                              _Float16* __restrict__ bcfrag) {
  int idx = blockIdx.x * blockDim.x + threadIdx.x;
  if (idx >= 2 * 4 * 32 * 16) return;
  int t  = idx & 15;
  int l  = (idx >> 4) & 31;
  int nb = (idx >> 9) & 3;
  int cb = idx >> 11;
  int i  = nb * 16 + (l & 15);
  int K  = ((t & 8) << 1) + ((l < 16) ? 0 : 8) + (t & 7);
  int j  = cb * 32 + K;
  bcfrag[idx] = (_Float16)be2[i * 64 + j];
}

// --- Transpose GRU weights [192,64] -> [64,192] for coalesced reads.
__global__ void wtrans_kernel(const float* __restrict__ Wih,
                              const float* __restrict__ Whh,
                              float* __restrict__ wihT,
                              float* __restrict__ whhT) {
  int idx = blockIdx.x * blockDim.x + threadIdx.x;
  if (idx >= 192 * 64) return;
  int r = idx >> 6, c = idx & 63;
  wihT[c * 192 + r] = Wih[idx];
  whhT[c * 192 + r] = Whh[idx];
}

// --- Edge encoder: eh = relu(ef @ We1^T + be1), stored f16 row-major [E,64].
__global__ void edge_enc_kernel(const float* __restrict__ ef,
                                const float* __restrict__ We1,
                                const float* __restrict__ be1,
                                _Float16* __restrict__ eh) {
  int idx = blockIdx.x * blockDim.x + threadIdx.x;
  if (idx >= N_EDGES * H) return;
  int e = idx >> 6, hh = idx & 63;
  float v = be1[hh];
#pragma unroll
  for (int k = 0; k < F_EDGE; ++k)
    v += ef[(size_t)e * F_EDGE + k] * We1[hh * F_EDGE + k];
  eh[idx] = (_Float16)(v > 0.0f ? v : 0.0f);
}

// --- Node init: x = nf @ Win^T + bin (f32 + f16 mirror).
__global__ void node_init_kernel(const float* __restrict__ nf,
                                 const float* __restrict__ Win,
                                 const float* __restrict__ binb,
                                 float* __restrict__ x,
                                 _Float16* __restrict__ xh) {
  int idx = blockIdx.x * blockDim.x + threadIdx.x;
  if (idx >= N_NODES * H) return;
  int n = idx >> 6, hh = idx & 63;
  float v = binb[hh];
#pragma unroll
  for (int k = 0; k < F_NODE; ++k)
    v += nf[(size_t)n * F_NODE + k] * Win[hh * F_NODE + k];
  x[idx]  = v;
  xh[idx] = (_Float16)v;
}

// --- Fused message kernel: one wave = 16 edges; 4 f32 accumulators cover
// msg[16 x 64]; 128 main K-chunks + 2 bias chunks; atomic scatter to agg.
__global__ void __launch_bounds__(256)
msg_kernel(const _Float16* __restrict__ eh,
           const _Float16* __restrict__ xh,
           const _Float16* __restrict__ wfrag,
           const _Float16* __restrict__ bcfrag,
           const int* __restrict__ Esrc,
           const int* __restrict__ Etgt,
           float* __restrict__ agg,
           int numTiles) {
  int wave = (blockIdx.x * blockDim.x + threadIdx.x) >> 5;
  if (wave >= numTiles) return;               // uniform per wave -> EXEC all-1
  int lane = threadIdx.x & 31;
  int m    = lane & 15;                        // A-matrix row (edge in tile)
  int kst  = (lane < 16) ? 0 : 8;              // per-lane K sub-offset (A layout)

  int e   = wave * 16 + m;
  int src = Esrc[e];

  // Preload the only 32 eh halves this lane ever needs (A-fragment halves).
  const _Float16* ehrow = eh + (size_t)e * H;
  const v8h e0 = *(const v8h*)(ehrow + kst);
  const v8h e1 = *(const v8h*)(ehrow + kst + 16);
  const v8h e2 = *(const v8h*)(ehrow + kst + 32);
  const v8h e3 = *(const v8h*)(ehrow + kst + 48);
  // Same pattern of the gathered x row (used for bias chunks + scalars).
  const _Float16* xrow = xh + (size_t)src * H;
  const v8h x0 = *(const v8h*)(xrow + kst);
  const v8h x1 = *(const v8h*)(xrow + kst + 16);
  const v8h x2 = *(const v8h*)(xrow + kst + 32);
  const v8h x3 = *(const v8h*)(xrow + kst + 48);

  v8f acc[4] = {};

#pragma unroll 2
  for (int c = 0; c < 128; ++c) {
    _Float16 s = xrow[c >> 1];                 // x[src][j], j = c/2
    v8h p1 = ((c & 1) ? e2 : e0) * splat8(s);  // on-the-fly outer product
    v8h p2 = ((c & 1) ? e3 : e1) * splat8(s);
    v16h a = cat8(p1, p2);
    const _Float16* wb = wfrag + ((size_t)c * 4 * 32) * 16;
    __builtin_prefetch(wb + 4 * 32 * 16 * 4, 0, 1);  // next chunks (L2-hot)
#pragma unroll
    for (int nb = 0; nb < 4; ++nb) {
      v16h b = *(const v16h*)(wb + ((size_t)(nb * 32 + lane)) * 16);
      acc[nb] = __builtin_amdgcn_wmma_f32_16x16x32_f16(
          false, a, false, b, (short)0, acc[nb], false, false);
    }
  }

  // be2 bias: msg += Bc @ x[src]  (2 chunks, A-operand = x fragments).
#pragma unroll
  for (int cb = 0; cb < 2; ++cb) {
    v16h a = cat8(cb ? x2 : x0, cb ? x3 : x1);
#pragma unroll
    for (int nb = 0; nb < 4; ++nb) {
      v16h b = *(const v16h*)(bcfrag + ((size_t)((cb * 4 + nb) * 32 + lane)) * 16);
      acc[nb] = __builtin_amdgcn_wmma_f32_16x16x32_f16(
          false, a, false, b, (short)0, acc[nb], false, false);
    }
  }

  // C/D layout: VGPR q -> row M = q (lanes 0-15) / q+8 (lanes 16-31), N = lane&15.
  int rbase = (lane < 16) ? 0 : 8;
  int tgt[8];
#pragma unroll
  for (int q = 0; q < 8; ++q) tgt[q] = Etgt[wave * 16 + rbase + q];
#pragma unroll
  for (int nb = 0; nb < 4; ++nb)
#pragma unroll
    for (int q = 0; q < 8; ++q)
      atomicAdd(agg + (size_t)tgt[q] * H + nb * 16 + (lane & 15), acc[nb][q]);
}

// --- GRU cell: 4 nodes per 256-thread block, rows staged in LDS.
__global__ void gru_kernel(const float* __restrict__ agg,
                           const float* __restrict__ x,
                           const float* __restrict__ wihT,
                           const float* __restrict__ bih,
                           const float* __restrict__ whhT,
                           const float* __restrict__ bhh,
                           float* __restrict__ xn,
                           _Float16* __restrict__ xnh) {
  __shared__ float sa[256];
  __shared__ float sx[256];
  int tid = threadIdx.x;
  int n0  = blockIdx.x * 4;
  sa[tid] = agg[(size_t)n0 * H + tid];
  sx[tid] = x[(size_t)n0 * H + tid];
  __syncthreads();
  int ln = tid >> 6, hh = tid & 63;
  float ir = bih[hh], iz = bih[64 + hh], inn = bih[128 + hh];
  float hr = bhh[hh], hz = bhh[64 + hh], hn = bhh[128 + hh];
  for (int c = 0; c < H; ++c) {
    float a = sa[ln * 64 + c];
    float h = sx[ln * 64 + c];
    const float* wi = wihT + c * 192;
    const float* wh = whhT + c * 192;
    ir  += a * wi[hh];       iz += a * wi[64 + hh];  inn += a * wi[128 + hh];
    hr  += h * wh[hh];       hz += h * wh[64 + hh];  hn  += h * wh[128 + hh];
  }
  float r  = 1.0f / (1.0f + __expf(-(ir + hr)));
  float z  = 1.0f / (1.0f + __expf(-(iz + hz)));
  float nn = tanhf(inn + r * hn);
  float hp = sx[ln * 64 + hh];
  float o  = (1.0f - z) * nn + z * hp;
  size_t oi = (size_t)(n0 + ln) * H + hh;
  xn[oi]  = o;
  xnh[oi] = (_Float16)o;
}

// --- Output head + graph pooling: out[n] = x@Wout^T + bout, segment-sum.
__global__ void out_kernel(const float* __restrict__ x,
                           const float* __restrict__ Wout,
                           const float* __restrict__ bout,
                           const int* __restrict__ batch,
                           float* __restrict__ out) {
  int n = blockIdx.x * blockDim.x + threadIdx.x;
  if (n >= N_NODES) return;
  float s = bout[0];
#pragma unroll
  for (int c = 0; c < H; ++c) s += x[(size_t)n * H + c] * Wout[c];
  atomicAdd(&out[batch[n]], s);
}

extern "C" void kernel_launch(void* const* d_in, const int* in_sizes, int n_in,
                              void* d_out, int out_size, void* d_ws, size_t ws_size,
                              hipStream_t stream) {
  const float* nf   = (const float*)d_in[0];
  const float* ef   = (const float*)d_in[1];
  const int*   Esrc = (const int*)d_in[2];
  const int*   Etgt = (const int*)d_in[3];
  const int*   batch= (const int*)d_in[4];
  const float* Win  = (const float*)d_in[5];
  const float* binb = (const float*)d_in[6];
  const float* We1  = (const float*)d_in[7];
  const float* be1  = (const float*)d_in[8];
  const float* We2  = (const float*)d_in[9];
  const float* be2  = (const float*)d_in[10];
  const float* Wih  = (const float*)d_in[11];
  const float* bih  = (const float*)d_in[12];
  const float* Whh  = (const float*)d_in[13];
  const float* bhh  = (const float*)d_in[14];
  const float* Wout = (const float*)d_in[15];
  const float* bout = (const float*)d_in[16];
  float* out = (float*)d_out;

  char* ws = (char*)d_ws;
  size_t off = 0;
  auto alloc = [&](size_t bytes) -> void* {
    void* p = ws + off;
    off += bytes;
    off = (off + 255) & ~(size_t)255;
    return p;
  };
  _Float16* eh     = (_Float16*)alloc((size_t)N_EDGES * H * 2);
  _Float16* wfrag  = (_Float16*)alloc((size_t)128 * 4 * 32 * 16 * 2);
  _Float16* bcfrag = (_Float16*)alloc((size_t)2 * 4 * 32 * 16 * 2);
  float*    x0     = (float*)alloc((size_t)N_NODES * H * 4);
  float*    x1     = (float*)alloc((size_t)N_NODES * H * 4);
  _Float16* xh0    = (_Float16*)alloc((size_t)N_NODES * H * 2);
  _Float16* xh1    = (_Float16*)alloc((size_t)N_NODES * H * 2);
  float*    agg    = (float*)alloc((size_t)N_NODES * H * 4);
  float*    wihT   = (float*)alloc((size_t)192 * 64 * 4);
  float*    whhT   = (float*)alloc((size_t)192 * 64 * 4);
  (void)ws_size; (void)in_sizes; (void)n_in;

  hipMemsetAsync(d_out, 0, (size_t)out_size * sizeof(float), stream);

  wfrag_kernel <<<(128 * 4 * 32 * 16 + 255) / 256, 256, 0, stream>>>(We2, wfrag);
  bcfrag_kernel<<<(2 * 4 * 32 * 16 + 255) / 256, 256, 0, stream>>>(be2, bcfrag);
  wtrans_kernel<<<(192 * 64 + 255) / 256, 256, 0, stream>>>(Wih, Whh, wihT, whhT);
  edge_enc_kernel<<<(N_EDGES * H + 255) / 256, 256, 0, stream>>>(ef, We1, be1, eh);
  node_init_kernel<<<(N_NODES * H + 255) / 256, 256, 0, stream>>>(nf, Win, binb, x0, xh0);

  float*    xc  = x0;  float*    xn  = x1;
  _Float16* xhc = xh0; _Float16* xhn = xh1;
  for (int t = 0; t < T_STEPS; ++t) {
    hipMemsetAsync(agg, 0, (size_t)N_NODES * H * sizeof(float), stream);
    msg_kernel<<<(N_TILES + 7) / 8, 256, 0, stream>>>(
        eh, xhc, wfrag, bcfrag, Esrc, Etgt, agg, N_TILES);
    gru_kernel<<<N_NODES / 4, 256, 0, stream>>>(
        agg, xc, wihT, bih, whhT, bhh, xn, xhn);
    float* tf = xc; xc = xn; xn = tf;
    _Float16* th = xhc; xhc = xhn; xhn = th;
  }

  out_kernel<<<(N_NODES + 255) / 256, 256, 0, stream>>>(xc, Wout, bout, batch, out);
}
